// GNNEncoder_71055938945160
// MI455X (gfx1250) — matrix operand order repaired and checked by reading, not compiled, not run on hardware
//
#include <hip/hip_runtime.h>
#include <hip/hip_bf16.h>
#include <math.h>

// ---------------------------------------------------------------------------
// CDNA5 / gfx1250 types
// ---------------------------------------------------------------------------
typedef __attribute__((ext_vector_type(16))) __bf16        v16bf;
typedef __attribute__((ext_vector_type(8)))  float         v8f;
typedef __attribute__((ext_vector_type(4)))  unsigned int  u32x4;
typedef __attribute__((ext_vector_type(8)))  int           i32x8;
typedef __attribute__((ext_vector_type(4)))  int           i32x4;

__device__ __forceinline__ float gelu_exact(float x) {
    return 0.5f * x * (1.0f + erff(x * 0.7071067811865475f));
}

// ---------------------------------------------------------------------------
// Generic zero kernel (works for f32 zeros and u32 zeros)
// ---------------------------------------------------------------------------
__global__ void zero_u32_kernel(unsigned* __restrict__ p, long n) {
    long t = (long)blockIdx.x * blockDim.x + threadIdx.x;
    if (t < n) p[t] = 0u;
}

// ---------------------------------------------------------------------------
// Cast + transpose weight:  W [K x NC] row-major f32  ->  Wt [NC x K] bf16
// ---------------------------------------------------------------------------
__global__ void transpose_cast_kernel(const float* __restrict__ W,
                                      __bf16* __restrict__ Wt,
                                      int K, int NC) {
    int idx = blockIdx.x * 256 + threadIdx.x;
    if (idx >= K * NC) return;
    int col = idx / K;
    int k   = idx - col * K;
    Wt[(long)col * K + k] = (__bf16)W[(long)k * NC + col];
}

// ---------------------------------------------------------------------------
// Expand per-head a[8,16,16] (h,d,e) into block-diagonal 128x128 bf16,
// stored column-major: Wt[col*128 + k], col = h*16+e, k = h*16+d.
// ---------------------------------------------------------------------------
__global__ void build_blockdiag_kernel(const float* __restrict__ a,
                                       __bf16* __restrict__ Wt) {
    int idx = blockIdx.x * 256 + threadIdx.x;
    if (idx >= 128 * 128) return;
    int col = idx >> 7;
    int k   = idx & 127;
    int hc  = col >> 4, hk = k >> 4;
    float v = (hc == hk) ? a[(hc * 16 + (k & 15)) * 16 + (col & 15)] : 0.0f;
    Wt[(long)col * 128 + k] = (__bf16)v;
}

// ---------------------------------------------------------------------------
// WMMA GEMM:  C[M x NC] = actOut( actIn(A[M x 128]) @ W + bias )
//   A  : f32, row stride lda;   Wt : bf16 column-major [NC][128]
// Block = 256 thr = 8 waves; wave w -> rows [blk*128 + 16w, +16), 128 cols.
// The 128x128 bf16 B panel (one contiguous 32KB slab of Wt) is staged in
// LDS by the Tensor Data Mover (one D# issued by wave 0, TENSORcnt wait,
// then workgroup barrier); the 8 waves start their A-fragment global
// loads + bf16 conversion after the barrier.  Falls back to a cooperative
// vector copy if the TDM builtin is unavailable.
// All M used here are multiples of 16, so the boundary guard is wave-uniform
// (EXEC stays all-ones inside WMMA as the ISA requires).
// ---------------------------------------------------------------------------
template <int ACT_IN_GELU, int ACT_OUT_RELU>
__global__ void gemm128_wmma_kernel(const float* __restrict__ A, long lda,
                                    const __bf16* __restrict__ Wt,
                                    const float* __restrict__ bias,
                                    float* __restrict__ C, long ldc,
                                    int M) {
    __shared__ __bf16 Bs[128 * 128];             // 32 KB of the 320 KB WGP LDS

    const int lane     = threadIdx.x & 31;
    const int wave     = threadIdx.x >> 5;
    const int rowBase  = blockIdx.x * 128 + wave * 16;
    const int colBase0 = blockIdx.y * 128;

#if __has_builtin(__builtin_amdgcn_tensor_load_to_lds) && \
    __has_builtin(__builtin_amdgcn_s_wait_tensorcnt)
    // ---- stage B panel via the Tensor Data Mover (async DMA to LDS) ----
    if (threadIdx.x < 32) {                      // one wave drives the TDM
        const unsigned ldsOff =
            (unsigned)(unsigned long long)(uintptr_t)&Bs[0];
        const unsigned long long ga =
            (unsigned long long)(uintptr_t)(Wt + (long)colBase0 * 128);
        // D# group 0: count=1 | lds_addr | global_addr[56:0] | type=2
        u32x4 g0;
        g0[0] = 1u;
        g0[1] = ldsOff;
        g0[2] = (unsigned)(ga & 0xFFFFFFFFu);
        g0[3] = (unsigned)((ga >> 32) & 0x01FFFFFFu) | (2u << 30);
        // D# group 1: data_size=4B; tensor_dim0=tile_dim0=8192 dwords;
        //             tensor_dim1=tile_dim1=1; dim0_stride=8192
        const unsigned nd = 8192u;               // 32KB / 4B
        i32x8 g1;
        g1[0] = (int)(2u << 16);                             // data_size=4B
        g1[1] = (int)((nd & 0xFFFFu) << 16);                 // tensor_dim0 lo
        g1[2] = (int)(((nd >> 16) & 0xFFFFu) | (1u << 16));  // dim0 hi|dim1 lo
        g1[3] = (int)((nd & 0xFFFFu) << 16);                 // dim1 hi|tile_dim0
        g1[4] = 1;                                           // tile_dim1=1
        g1[5] = (int)nd;                                     // dim0_stride lo
        g1[6] = 0;
        g1[7] = 0;
        i32x4 gz4; gz4[0] = gz4[1] = gz4[2] = gz4[3] = 0;    // groups 2/3 (<=2D)
        i32x8 gz8;
        #pragma unroll
        for (int i = 0; i < 8; ++i) gz8[i] = 0;
        // 6-arg form (clang-23 / therock-10.0 headers)
        __builtin_amdgcn_tensor_load_to_lds(g0, g1, gz4, gz4, gz8, 0);
        __builtin_amdgcn_s_wait_tensorcnt(0);
    }
#else
    // ---- fallback: cooperative copy of the contiguous B panel ----
    {
        const uint4* srcp = (const uint4*)(Wt + (long)colBase0 * 128);
        uint4*       dstp = (uint4*)Bs;          // 2048 x 16B chunks
        #pragma unroll
        for (int i = 0; i < 8; ++i)
            dstp[threadIdx.x + i * 256] = srcp[threadIdx.x + i * 256];
    }
#endif
    __syncthreads();

    if (rowBase >= M) return;                    // uniform per wave

    const int halfSel = lane >> 4;               // 0 | 1
    const int lm      = lane & 15;
    const int row     = rowBase + lm;
    const float* arow = A + (long)row * lda;

    v8f acc[8];
    #pragma unroll
    for (int t = 0; t < 8; ++t)
        #pragma unroll
        for (int j = 0; j < 8; ++j) acc[t][j] = 0.0f;

    #pragma unroll
    for (int kk = 0; kk < 128; kk += 32) {
        __builtin_prefetch(arow + kk + 32, 0, 3);   // global_prefetch_b8
        // ---- A fragment (contiguous k order: k = 16*halfSel + s, matches B) ----
        v16bf afrag;
        const int o = halfSel * 16;
        #pragma unroll
        for (int i = 0; i < 8; ++i) {
            float f0 = arow[kk + o + i];
            float f1 = arow[kk + o + 8 + i];
            if (ACT_IN_GELU) { f0 = gelu_exact(f0); f1 = gelu_exact(f1); }
            afrag[i]     = (__bf16)f0;
            afrag[8 + i] = (__bf16)f1;
        }
        // ---- all 8 B fragments (ds_load_b128 pairs), then 8 WMMAs ----
        v16bf bfr[8];
        #pragma unroll
        for (int t = 0; t < 8; ++t)
            bfr[t] = *(const v16bf*)(Bs + (long)(t * 16 + lm) * 128 + kk +
                                     halfSel * 16);
        #pragma unroll
        for (int t = 0; t < 8; ++t)
            acc[t] = __builtin_amdgcn_wmma_f32_16x16x32_bf16(
                false, afrag, false, bfr[t], (short)0, acc[t], false, false);
    }

    // ---- store (ISA f32 C/D layout: VGPR j -> row j + 8*halfSel) ----
    #pragma unroll
    for (int t = 0; t < 8; ++t) {
        const int col = colBase0 + t * 16 + lm;
        const float b = bias ? bias[col] : 0.0f;
        #pragma unroll
        for (int j = 0; j < 8; ++j) {
            const int r = rowBase + j + halfSel * 8;
            float v = acc[t][j] + b;
            if (ACT_OUT_RELU) v = fmaxf(v, 0.0f);
            C[(long)r * ldc + col] = v;
        }
    }
}

// ---------------------------------------------------------------------------
// Edge phase.  One thread per (edge, head).
// ---------------------------------------------------------------------------
__device__ __forceinline__ unsigned f2key(float f) {
    unsigned u = __float_as_uint(f);
    return (u & 0x80000000u) ? ~u : (u | 0x80000000u);
}
__device__ __forceinline__ float key2f(unsigned k) {
    unsigned u = (k & 0x80000000u) ? (k ^ 0x80000000u) : ~k;
    return __uint_as_float(u);
}

__global__ void attn_logits_kernel(const int* __restrict__ src,
                                   const int* __restrict__ dst, int E,
                                   const float* __restrict__ kqv_dst, // [Nd,384]
                                   const float* __restrict__ ke,      // [Ns,128]
                                   const float* __restrict__ prel,    // [8]
                                   float* __restrict__ alpha,         // [E,8]
                                   unsigned* __restrict__ mkeys) {    // [Nd,8]
    int t = blockIdx.x * 256 + threadIdx.x;
    if (t >= E * 8) return;
    int e = t >> 3, h = t & 7;
    int s = src[e], d = dst[e];
    const float* qp = kqv_dst + (long)d * 384 + 128 + h * 16;
    const float* kp = ke      + (long)s * 128 + h * 16;
    float acc = 0.0f;
    #pragma unroll
    for (int i = 0; i < 16; ++i) acc += qp[i] * kp[i];
    float a = acc * prel[h] * 0.25f;              // scale = 1/sqrt(16)
    alpha[t] = a;
    atomicMax(&mkeys[(long)d * 8 + h], f2key(a));
}

__global__ void attn_exp_kernel(const int* __restrict__ dst, int E,
                                float* __restrict__ alpha,
                                const unsigned* __restrict__ mkeys,
                                float* __restrict__ ssum) {
    int t = blockIdx.x * 256 + threadIdx.x;
    if (t >= E * 8) return;
    int e = t >> 3, h = t & 7;
    int d = dst[e];
    float m  = key2f(mkeys[(long)d * 8 + h]);
    float ex = expf(alpha[t] - m);
    alpha[t] = ex;
    unsafeAtomicAdd(&ssum[(long)d * 8 + h], ex);
}

__global__ void attn_agg_kernel(const int* __restrict__ src,
                                const int* __restrict__ dst, int E,
                                const float* __restrict__ ve,     // [Ns,128]
                                const float* __restrict__ alpha,  // [E,8]
                                const float* __restrict__ ssum,   // [Nd,8]
                                float* __restrict__ agg) {        // [Nd,128]
    int t = blockIdx.x * 256 + threadIdx.x;
    if (t >= E * 8) return;
    int e = t >> 3, h = t & 7;
    int s = src[e], d = dst[e];
    float w = alpha[t] / (ssum[(long)d * 8 + h] + 1e-16f);
    const float* vp = ve + (long)s * 128 + h * 16;
    float* ap = agg + (long)d * 128 + h * 16;
    #pragma unroll
    for (int i = 0; i < 16; ++i) unsafeAtomicAdd(&ap[i], vp[i] * w);
}

// ---------------------------------------------------------------------------
// Skip connection + column sum/sumsq (block handles 64 rows, private accum,
// two atomics (one per 128-thread half) per column per block).
// ---------------------------------------------------------------------------
__global__ void skip_stats_kernel(const float* __restrict__ outp,
                                  const float* __restrict__ xin,
                                  const float* __restrict__ skipP,
                                  float* __restrict__ hbuf,
                                  float* __restrict__ colsum,
                                  float* __restrict__ colsumsq, int N) {
    const int col  = threadIdx.x & 127;
    const int half = threadIdx.x >> 7;
    const int rowBase = blockIdx.x * 64;
    const float sk = 1.0f / (1.0f + expf(-skipP[0]));
    float s0 = 0.0f, s1 = 0.0f;
    for (int i = 0; i < 32; ++i) {
        int r = rowBase + half + i * 2;
        if (r < N) {
            long idx = (long)r * 128 + col;
            float v = sk * outp[idx] + (1.0f - sk) * xin[idx];
            hbuf[idx] = v;
            s0 += v; s1 += v * v;
        }
    }
    unsafeAtomicAdd(&colsum[col], s0);
    unsafeAtomicAdd(&colsumsq[col], s1);
}

__global__ void bn_apply_kernel(const float* __restrict__ hbuf,
                                const float* __restrict__ colsum,
                                const float* __restrict__ colsumsq,
                                const float* __restrict__ gamma,
                                const float* __restrict__ beta,
                                float* __restrict__ xout, int N) {
    long t = (long)blockIdx.x * 256 + threadIdx.x;
    if (t >= (long)N * 128) return;
    int col = (int)(t & 127);
    float invN = 1.0f / (float)N;
    float mu  = colsum[col] * invN;
    float var = colsumsq[col] * invN - mu * mu;
    xout[t] = (hbuf[t] - mu) * rsqrtf(var + 1e-5f) * gamma[col] + beta[col];
}

// ---------------------------------------------------------------------------
// Host-side orchestration
// ---------------------------------------------------------------------------
static inline size_t alignUp256(size_t x) { return (x + 255) & ~(size_t)255; }
static inline int cdiv(long a, long b) { return (int)((a + b - 1) / b); }

extern "C" void kernel_launch(void* const* d_in, const int* in_sizes, int n_in,
                              void* d_out, int out_size, void* d_ws, size_t ws_size,
                              hipStream_t stream) {
    (void)n_in; (void)out_size; (void)ws_size;

    const float* x_author     = (const float*)d_in[0];
    const float* x_paper      = (const float*)d_in[1];
    const int*   writes_src   = (const int*)d_in[2];
    const int*   writes_dst   = (const int*)d_in[3];
    const int*   rev_src      = (const int*)d_in[4];
    const int*   rev_dst      = (const int*)d_in[5];
    const int*   cites_src    = (const int*)d_in[6];
    const int*   cites_dst    = (const int*)d_in[7];
    const float* lin_W[2]     = {(const float*)d_in[8], (const float*)d_in[10]};
    const float* lin_b[2]     = {(const float*)d_in[9], (const float*)d_in[11]};
    const float* kqv_W        = (const float*)d_in[12];
    const float* kqv_b        = (const float*)d_in[13];
    const float* a_k          = (const float*)d_in[14];
    const float* a_v          = (const float*)d_in[15];
    const float* p_rel        = (const float*)d_in[16];
    const float* out_W        = (const float*)d_in[17];
    const float* out_b        = (const float*)d_in[18];
    const float* skipP        = (const float*)d_in[19];
    const float* bn_gamma     = (const float*)d_in[20];
    const float* bn_beta      = (const float*)d_in[21];

    const int NA = in_sizes[0] / 128;
    const int NP = in_sizes[1] / 128;
    const int EW = in_sizes[2], ER = in_sizes[4], EC = in_sizes[6];
    const int Nt[2] = {NA, NP};
    const int L = 2;

    // ---- workspace carve-out ----
    char* ws = (char*)d_ws;
    size_t off = 0;
    auto alloc = [&](size_t bytes) -> void* {
        void* p = ws + off;
        off = alignUp256(off + bytes);
        return p;
    };
    float* xcur[2]; xcur[0] = (float*)alloc((size_t)NA*128*4); xcur[1] = (float*)alloc((size_t)NP*128*4);
    float* kqvB[2]; kqvB[0] = (float*)alloc((size_t)NA*384*4); kqvB[1] = (float*)alloc((size_t)NP*384*4);
    const int relS[3] = {0, 1, 1}, relD[3] = {1, 0, 1};
    const int relE[3] = {EW, ER, EC};
    const int* relSrc[3] = {writes_src, rev_src, cites_src};
    const int* relDst[3] = {writes_dst, rev_dst, cites_dst};
    float* keB[3]; float* veB[3]; float* alB[3];
    for (int r = 0; r < 3; ++r) {
        keB[r] = (float*)alloc((size_t)Nt[relS[r]]*128*4);
        veB[r] = (float*)alloc((size_t)Nt[relS[r]]*128*4);
        alB[r] = (float*)alloc((size_t)relE[r]*8*4);
    }
    unsigned* mkeys[2]; float* ssum[2]; float* agg[2]; float* outB[2];
    for (int t = 0; t < 2; ++t) {
        mkeys[t] = (unsigned*)alloc((size_t)Nt[t]*8*4);
        ssum[t]  = (float*)alloc((size_t)Nt[t]*8*4);
        agg[t]   = (float*)alloc((size_t)Nt[t]*128*4);
        outB[t]  = (float*)alloc((size_t)Nt[t]*128*4);
    }
    float* colsum   = (float*)alloc(128*4);
    float* colsumsq = (float*)alloc(128*4);
    __bf16* linT[2]; linT[0] = (__bf16*)alloc(128*128*2); linT[1] = (__bf16*)alloc(128*128*2);
    __bf16* kqvT = (__bf16*)alloc((size_t)L*2*128*384*2);
    __bf16* outT = (__bf16*)alloc((size_t)L*2*128*128*2);
    __bf16* bdK  = (__bf16*)alloc((size_t)L*3*128*128*2);
    __bf16* bdV  = (__bf16*)alloc((size_t)L*3*128*128*2);

    // ---- weight prep (bf16, column-major / block-diagonal) ----
    for (int t = 0; t < 2; ++t)
        transpose_cast_kernel<<<cdiv(128*128,256),256,0,stream>>>(lin_W[t], linT[t], 128, 128);
    for (int l = 0; l < L; ++l)
        for (int t = 0; t < 2; ++t) {
            transpose_cast_kernel<<<cdiv(128*384,256),256,0,stream>>>(
                kqv_W + (size_t)(l*2+t)*128*384, kqvT + (size_t)(l*2+t)*384*128, 128, 384);
            transpose_cast_kernel<<<cdiv(128*128,256),256,0,stream>>>(
                out_W + (size_t)(l*2+t)*128*128, outT + (size_t)(l*2+t)*128*128, 128, 128);
        }
    for (int l = 0; l < L; ++l)
        for (int r = 0; r < 3; ++r) {
            build_blockdiag_kernel<<<cdiv(128*128,256),256,0,stream>>>(
                a_k + (size_t)(l*3+r)*8*16*16, bdK + (size_t)(l*3+r)*128*128);
            build_blockdiag_kernel<<<cdiv(128*128,256),256,0,stream>>>(
                a_v + (size_t)(l*3+r)*8*16*16, bdV + (size_t)(l*3+r)*128*128);
        }

    // ---- input projection + ReLU ----
    gemm128_wmma_kernel<0,1><<<dim3(cdiv(NA,128),1),256,0,stream>>>(
        x_author, 128, linT[0], lin_b[0], xcur[0], 128, NA);
    gemm128_wmma_kernel<0,1><<<dim3(cdiv(NP,128),1),256,0,stream>>>(
        x_paper, 128, linT[1], lin_b[1], xcur[1], 128, NP);

    // ---- HGT layers ----
    for (int l = 0; l < L; ++l) {
        // K/Q/V projection per node type
        for (int t = 0; t < 2; ++t)
            gemm128_wmma_kernel<0,0><<<dim3(cdiv(Nt[t],128),3),256,0,stream>>>(
                xcur[t], 128, kqvT + (size_t)(l*2+t)*384*128,
                kqv_b + (size_t)(l*2+t)*384, kqvB[t], 384, Nt[t]);

        // per-relation K/V transforms (block-diagonal WMMA GEMM)
        for (int r = 0; r < 3; ++r) {
            int s = relS[r];
            gemm128_wmma_kernel<0,0><<<dim3(cdiv(Nt[s],128),1),256,0,stream>>>(
                kqvB[s] + 0,   384, bdK + (size_t)(l*3+r)*128*128, nullptr, keB[r], 128, Nt[s]);
            gemm128_wmma_kernel<0,0><<<dim3(cdiv(Nt[s],128),1),256,0,stream>>>(
                kqvB[s] + 256, 384, bdV + (size_t)(l*3+r)*128*128, nullptr, veB[r], 128, Nt[s]);
        }

        // reset segment-max keys, denominators, aggregation buffers
        for (int t = 0; t < 2; ++t) {
            zero_u32_kernel<<<cdiv((long)Nt[t]*8,256),256,0,stream>>>(mkeys[t], (long)Nt[t]*8);
            zero_u32_kernel<<<cdiv((long)Nt[t]*8,256),256,0,stream>>>((unsigned*)ssum[t], (long)Nt[t]*8);
            zero_u32_kernel<<<cdiv((long)Nt[t]*128,256),256,0,stream>>>((unsigned*)agg[t], (long)Nt[t]*128);
        }

        // attention: logits + segment max
        for (int r = 0; r < 3; ++r)
            attn_logits_kernel<<<cdiv((long)relE[r]*8,256),256,0,stream>>>(
                relSrc[r], relDst[r], relE[r], kqvB[relD[r]], keB[r],
                p_rel + (size_t)(l*3+r)*8, alB[r], mkeys[relD[r]]);
        // exp + segment sum
        for (int r = 0; r < 3; ++r)
            attn_exp_kernel<<<cdiv((long)relE[r]*8,256),256,0,stream>>>(
                relDst[r], relE[r], alB[r], mkeys[relD[r]], ssum[relD[r]]);
        // weighted aggregation
        for (int r = 0; r < 3; ++r)
            attn_agg_kernel<<<cdiv((long)relE[r]*8,256),256,0,stream>>>(
                relSrc[r], relDst[r], relE[r], veB[r], alB[r], ssum[relD[r]], agg[relD[r]]);

        // output projection (GELU fused into A conversion), skip, BatchNorm
        for (int t = 0; t < 2; ++t) {
            gemm128_wmma_kernel<1,0><<<dim3(cdiv(Nt[t],128),1),256,0,stream>>>(
                agg[t], 128, outT + (size_t)(l*2+t)*128*128,
                out_b + (size_t)(l*2+t)*128, outB[t], 128, Nt[t]);

            zero_u32_kernel<<<1,256,0,stream>>>((unsigned*)colsum, 128);
            zero_u32_kernel<<<1,256,0,stream>>>((unsigned*)colsumsq, 128);
            skip_stats_kernel<<<cdiv(Nt[t],64),256,0,stream>>>(
                outB[t], xcur[t], skipP + (size_t)(l*2+t),
                outB[t] /* h in-place */, colsum, colsumsq, Nt[t]);

            float* dst = (l == L-1)
                ? ((float*)d_out + (t == 0 ? 0 : (size_t)NA*128))
                : xcur[t];
            bn_apply_kernel<<<cdiv((long)Nt[t]*128,256),256,0,stream>>>(
                outB[t], colsum, colsumsq, bn_gamma + (size_t)l*128,
                bn_beta + (size_t)l*128, dst, Nt[t]);
        }
    }
}
